// TernaryLinear_64639257805186
// MI455X (gfx1250) — matrix operand order repaired and checked by reading, not compiled, not run on hardware
//
#include <hip/hip_runtime.h>

typedef __bf16 bf16;
typedef __attribute__((ext_vector_type(16))) __bf16       v16bf;
typedef __attribute__((ext_vector_type(8)))  __bf16       v8bf;
typedef __attribute__((ext_vector_type(8)))  float        v8f;
typedef __attribute__((ext_vector_type(4)))  unsigned int v4u;
typedef __attribute__((ext_vector_type(8)))  int          v8i;
typedef __attribute__((ext_vector_type(4)))  int          v4i;

#define IN_F  512
#define OUT_F 512
#define EPS   1e-4f

// B panel in LDS: 64 rows (columns of tw), each 512 bf16 = 1024 B, padded by the
// TDM with 4 DWORDs (16 B) every 256 DWORDs -> row stride 1040 B.
// 1040/4 = 260 dwords, 260 % 64 = 4 -> row r starts at bank 4*r: 16 consecutive
// rows hit 16 distinct bank groups (conflict-free N-tile reads).
#define LDS_ROW_BYTES 1040
#define LDS_B_BYTES   (64 * LDS_ROW_BYTES)

#if __has_builtin(__builtin_amdgcn_tensor_load_to_lds) && \
    __has_builtin(__builtin_amdgcn_s_wait_tensorcnt)
#define HAVE_TDM 1
#else
#define HAVE_TDM 0
#endif

// ---------------------------------------------------------------------------
// Phase 1: ternarize W (per-row delta/alpha), write twT[j*IN + i] = alpha_i*t[i,j]
// transposed + bf16 so the GEMM B-operand is a contiguous K-run per lane.
// ---------------------------------------------------------------------------
__global__ __launch_bounds__(256)
void ternarize_rows(const float* __restrict__ W, bf16* __restrict__ twT) {
    __shared__ float s_sum[256];
    __shared__ int   s_cnt[256];

    const int i   = blockIdx.x;          // row of W (input-feature index)
    const int tid = threadIdx.x;

    const float w0 = W[(size_t)i * OUT_F + tid];
    const float w1 = W[(size_t)i * OUT_F + tid + 256];
    const float a0 = fabsf(w0);
    const float a1 = fabsf(w1);

    s_sum[tid] = a0 + a1;
    __syncthreads();
    for (int s = 128; s > 0; s >>= 1) {
        if (tid < s) s_sum[tid] += s_sum[tid + s];
        __syncthreads();
    }
    const float delta = (0.7f / (float)OUT_F) * s_sum[0];
    __syncthreads();

    const bool m0 = a0 > delta;
    const bool m1 = a1 > delta;

    s_sum[tid] = (m0 ? a0 : 0.0f) + (m1 ? a1 : 0.0f);
    s_cnt[tid] = (m0 ? 1 : 0) + (m1 ? 1 : 0);
    __syncthreads();
    for (int s = 128; s > 0; s >>= 1) {
        if (tid < s) { s_sum[tid] += s_sum[tid + s]; s_cnt[tid] += s_cnt[tid + s]; }
        __syncthreads();
    }
    int   cnt   = s_cnt[0]; if (cnt == 0) cnt = 1;
    float alpha = s_sum[0] / (float)cnt;
    alpha = fmaxf(alpha, EPS);

    const float t0 = m0 ? (w0 > 0.0f ? alpha : -alpha) : 0.0f;
    const float t1 = m1 ? (w1 > 0.0f ? alpha : -alpha) : 0.0f;

    twT[(size_t)(tid)       * IN_F + i] = (bf16)t0;
    twT[(size_t)(tid + 256) * IN_F + i] = (bf16)t1;
}

// ---------------------------------------------------------------------------
// A operand loader: 16-bit A 16x32 ISA layout.
// lanes 0-15 hold K = k0+{0..7, 16..23}; lanes 16-31 hold K = k0+{8..15, 24..31}.
// ---------------------------------------------------------------------------
__device__ __forceinline__ v16bf load_a(const float* __restrict__ xrow, int k0, int half) {
    const int ka = k0 + half * 8;
    const float4 f0 = *(const float4*)(xrow + ka);
    const float4 f1 = *(const float4*)(xrow + ka + 4);
    const float4 f2 = *(const float4*)(xrow + ka + 16);
    const float4 f3 = *(const float4*)(xrow + ka + 20);
    v16bf A;
    A[0]  = (bf16)f0.x; A[1]  = (bf16)f0.y; A[2]  = (bf16)f0.z; A[3]  = (bf16)f0.w;
    A[4]  = (bf16)f1.x; A[5]  = (bf16)f1.y; A[6]  = (bf16)f1.z; A[7]  = (bf16)f1.w;
    A[8]  = (bf16)f2.x; A[9]  = (bf16)f2.y; A[10] = (bf16)f2.z; A[11] = (bf16)f2.w;
    A[12] = (bf16)f3.x; A[13] = (bf16)f3.y; A[14] = (bf16)f3.z; A[15] = (bf16)f3.w;
    return A;
}

// ---------------------------------------------------------------------------
// Phase 2: out = x @ twT^T + b via v_wmma_f32_16x16x32_bf16.
// Block = 256 threads = 8 waves; block tile = 256 M x 64 N.
// Wave tile = 32 M x 64 N: 2 A operands, 4 B operands, 8 WMMAs per K-step.
// B panel staged global->LDS once per block by the Tensor Data Mover.
// ---------------------------------------------------------------------------
__global__ __launch_bounds__(256)
void ternary_gemm_wmma(const float* __restrict__ X,
                       const bf16*  __restrict__ twT,
                       const float* __restrict__ bias,
                       float*       __restrict__ out) {
    __shared__ __attribute__((aligned(64))) unsigned char sB[LDS_B_BYTES];

    const int lane = threadIdx.x & 31;
    const int wave = threadIdx.x >> 5;
    const int half = lane >> 4;
    const int l16  = lane & 15;

    const int m0 = blockIdx.x * 256 + wave * 32;
    const int n0 = blockIdx.y * 64;

#if HAVE_TDM
    // ---- TDM: DMA the 64x512 bf16 B panel into LDS (one op, wave 0 only).
    if (wave == 0) {
        const unsigned long long gaddr =
            (unsigned long long)(const void*)(twT + (size_t)n0 * IN_F);
        const unsigned int laddr =
            (unsigned int)(unsigned long long)(const void*)sB; // low 32b = LDS offset

        v4u g0;
        g0[0] = 1u;                                       // count=1 (valid user D#)
        g0[1] = laddr;                                    // lds_addr
        g0[2] = (unsigned int)gaddr;                      // global_addr[31:0]
        g0[3] = ((unsigned int)(gaddr >> 32) & 0x01FFFFFFu) // global_addr[56:32]
                | 0x80000000u;                            // type = 2 ("image")

        v8i g1;
        g1[0] = (int)((1u << 16)      // data_size = 1 -> 2 bytes
                    | (1u << 20)      // pad_enable
                    | (7u << 22)      // pad_interval code 7 -> every 256 DWORDs
                    | (3u << 25));    // pad_amount  code 3 -> 4 DWORDs (16 B)
        g1[1] = (int)(512u << 16);    // tensor_dim0 = 512 (elems)
        g1[2] = (int)(64u  << 16);    // tensor_dim1 = 64
        g1[3] = (int)(512u << 16);    // tile_dim0 = 512
        g1[4] = 64;                   // tile_dim1 = 64 (tile_dim2 = 0)
        g1[5] = 512;                  // tensor_dim0_stride = 512 (elems)
        g1[6] = 0;
        g1[7] = 0;

        v4i gz = {0, 0, 0, 0};        // 2-D tensor: groups 2/3 unused
#if defined(__clang_major__) && (__clang_major__ >= 23)
        v8i gz8 = {0, 0, 0, 0, 0, 0, 0, 0};
        __builtin_amdgcn_tensor_load_to_lds(g0, g1, gz, gz, gz8, 0);
#else
        __builtin_amdgcn_tensor_load_to_lds(g0, g1, gz, gz, 0);
#endif
        __builtin_amdgcn_s_wait_tensorcnt(0);
    }
#else
    // ---- fallback: cooperative synchronous copy into the same padded layout.
    for (int idx = threadIdx.x; idx < 64 * (IN_F / 8); idx += 256) {
        const int row   = idx >> 6;     // 64 16-byte chunks per row
        const int chunk = idx & 63;
        v8bf v = *(const v8bf*)(twT + (size_t)(n0 + row) * IN_F + chunk * 8);
        *(v8bf*)(sB + row * LDS_ROW_BYTES + chunk * 16) = v;
    }
#endif
    __syncthreads();

    // ---- accumulators, bias-initialized (C layout: all 8 regs share N = l16)
    v8f acc[2][4];
#pragma unroll
    for (int t = 0; t < 4; ++t) {
        const float bv = bias[n0 + t * 16 + l16];
        v8f c;
#pragma unroll
        for (int k = 0; k < 8; ++k) c[k] = bv;
        acc[0][t] = c;
        acc[1][t] = c;
    }

    const float* xrow0 = X + (size_t)(m0 + l16) * IN_F;
    const float* xrow1 = xrow0 + (size_t)16 * IN_F;

    for (int k0 = 0; k0 < IN_F; k0 += 32) {
        const v16bf A0 = load_a(xrow0, k0, half);
        const v16bf A1 = load_a(xrow1, k0, half);

        // B: 32x16 bf16; lane = column n, lanes 0-15 K=k0+{0..7,16..23}, 16-31 K=k0+{8..15,24..31}.
#pragma unroll
        for (int t = 0; t < 4; ++t) {
            const unsigned char* p =
                sB + (t * 16 + l16) * LDS_ROW_BYTES + (k0 + half * 8) * 2;
            const v8bf b0 = *(const v8bf*)(p);
            const v8bf b1 = *(const v8bf*)(p + 32);
            const v16bf B = __builtin_shufflevector(
                b0, b1, 0, 1, 2, 3, 4, 5, 6, 7, 8, 9, 10, 11, 12, 13, 14, 15);

            acc[0][t] = __builtin_amdgcn_wmma_f32_16x16x32_bf16(
                false, A0, false, B, (short)0, acc[0][t], false, false);
            acc[1][t] = __builtin_amdgcn_wmma_f32_16x16x32_bf16(
                false, A1, false, B, (short)0, acc[1][t], false, false);
        }
    }

    // ---- store: row = m0 + mt*16 + v + 8*half, col = n0 + t*16 + l16
#pragma unroll
    for (int mt = 0; mt < 2; ++mt) {
#pragma unroll
        for (int t = 0; t < 4; ++t) {
            const int n = n0 + t * 16 + l16;
#pragma unroll
            for (int v = 0; v < 8; ++v) {
                const int m = m0 + mt * 16 + v + half * 8;
                out[(size_t)m * OUT_F + n] = acc[mt][t][v];
            }
        }
    }
}

extern "C" void kernel_launch(void* const* d_in, const int* in_sizes, int n_in,
                              void* d_out, int out_size, void* d_ws, size_t ws_size,
                              hipStream_t stream) {
    const float* x = (const float*)d_in[0];
    const float* W = (const float*)d_in[1];
    const float* b = (const float*)d_in[2];
    float* out = (float*)d_out;

    bf16* twT = (bf16*)d_ws;            // 512 KB workspace

    const int N = in_sizes[0] / IN_F;   // 131072

    ternarize_rows<<<IN_F, 256, 0, stream>>>(W, twT);

    dim3 grid(N / 256, OUT_F / 64);     // (512, 8)
    ternary_gemm_wmma<<<grid, 256, 0, stream>>>(x, twT, b, out);
}
